// ContraNorm_19035295056148
// MI455X (gfx1250) — matrix operand order repaired and checked by reading, not compiled, not run on hardware
//
#include <hip/hip_runtime.h>
#include <cstdint>
#include <cstddef>

// ContraNorm for x[8192,512] f32 on gfx1250 (MI455X), wave32 + WMMA bf16.
//   sim = xn@xn^T is symmetric and bounded in [-1,1]  ->  softmax needs no max.
//   dual softmax:  x_neg = diag(1/s) * (E@x) + E @ (x/s),  E = exp(sim), s = row sums.
// Pipeline: prep (norms, bf16 casts) -> row-sum flash sweep (WMMA) ->
//           u = x/s -> fused recompute+second-GEMM+LayerNorm (WMMA).

#define N_ROWS 8192
#define D_DIM  512
#define SCALE_F 0.1f
#define LN_EPS 1e-6f
#define NRM_EPS 1e-12f
#define JT 256
#define JTP 264   // padded LDS row (halves), multiple of 8 -> 16B aligned rows

typedef __attribute__((ext_vector_type(16))) __bf16 v16bf;
typedef __attribute__((ext_vector_type(8)))  float  v8f;

union VHalf16 { v16bf v; uint4 q[2]; };

__device__ __forceinline__ unsigned short f2bf(float f) {
    // round-to-nearest-even f32 -> bf16 (inputs are finite)
    unsigned int u = __float_as_uint(f);
    u += 0x7fffu + ((u >> 16) & 1u);
    return (unsigned short)(u >> 16);
}

// ---------------------------------------------------------------------------
// Kernel 1: per-row L2 norm; write xn (bf16, row-major) and x^T (bf16).
// ---------------------------------------------------------------------------
__global__ void __launch_bounds__(128) k_prep(const float* __restrict__ x,
                                              unsigned short* __restrict__ xn,
                                              unsigned short* __restrict__ xbT) {
    const int row = blockIdx.x;
    const int tid = threadIdx.x;              // 128 threads = 4 waves
    const float* xr = x + (size_t)row * D_DIM;
    float v0 = xr[tid], v1 = xr[tid + 128], v2 = xr[tid + 256], v3 = xr[tid + 384];
    float ss = v0 * v0 + v1 * v1 + v2 * v2 + v3 * v3;
    #pragma unroll
    for (int off = 16; off > 0; off >>= 1) ss += __shfl_xor(ss, off, 32);
    __shared__ float part[4];
    if ((tid & 31) == 0) part[tid >> 5] = ss;
    __syncthreads();
    const float tot = part[0] + part[1] + part[2] + part[3];
    const float inv = 1.0f / fmaxf(sqrtf(tot), NRM_EPS);
    unsigned short* xnr = xn + (size_t)row * D_DIM;
    xnr[tid]       = f2bf(v0 * inv);
    xnr[tid + 128] = f2bf(v1 * inv);
    xnr[tid + 256] = f2bf(v2 * inv);
    xnr[tid + 384] = f2bf(v3 * inv);
    xbT[(size_t)(tid)        * N_ROWS + row] = f2bf(v0);
    xbT[(size_t)(tid + 128)  * N_ROWS + row] = f2bf(v1);
    xbT[(size_t)(tid + 256)  * N_ROWS + row] = f2bf(v2);
    xbT[(size_t)(tid + 384)  * N_ROWS + row] = f2bf(v3);
}

// ---------------------------------------------------------------------------
// Kernel 2: s_i = sum_j exp(xn_i . xn_j) via WMMA sweep (no max needed,
// logits bounded in [-1,1]). Block = 4 waves, 64 rows; wave owns 16 rows.
// ---------------------------------------------------------------------------
__global__ void __launch_bounds__(128) k_rowsum(const unsigned short* __restrict__ xn,
                                                float* __restrict__ srow) {
    const int lane  = threadIdx.x & 31;
    const int wave  = threadIdx.x >> 5;          // 0..3
    const int lr    = lane & 15;
    const int abase = (lane < 16) ? 0 : 16;      // A-operand K base (halves)
    const int kboff = (lane < 16) ? 0 : 16;      // B-operand K base (halves)
    const int i0    = blockIdx.x * 64 + wave * 16;

    const unsigned short* arow = xn + (size_t)(i0 + lr) * D_DIM;
    float sum[8];
    #pragma unroll
    for (int k = 0; k < 8; ++k) sum[k] = 0.f;

    for (int j = 0; j < N_ROWS; j += 32) {
        const unsigned short* b0row = xn + (size_t)(j + lr) * D_DIM;
        const unsigned short* b1row = xn + (size_t)(j + 16 + lr) * D_DIM;
        v8f acc0 = {0.f,0.f,0.f,0.f,0.f,0.f,0.f,0.f};
        v8f acc1 = {0.f,0.f,0.f,0.f,0.f,0.f,0.f,0.f};
        #pragma unroll
        for (int kk = 0; kk < 16; ++kk) {
            const int kb = kk * 32;
            VHalf16 a, b0, b1;
            a.q[0]  = *(const uint4*)(arow  + kb + abase);
            a.q[1]  = *(const uint4*)(arow  + kb + abase + 8);
            b0.q[0] = *(const uint4*)(b0row + kb + kboff);
            b0.q[1] = *(const uint4*)(b0row + kb + kboff + 8);
            b1.q[0] = *(const uint4*)(b1row + kb + kboff);
            b1.q[1] = *(const uint4*)(b1row + kb + kboff + 8);
            acc0 = __builtin_amdgcn_wmma_f32_16x16x32_bf16(false, a.v, false, b0.v,
                                                           (short)0, acc0, false, false);
            acc1 = __builtin_amdgcn_wmma_f32_16x16x32_bf16(false, a.v, false, b1.v,
                                                           (short)0, acc1, false, false);
        }
        #pragma unroll
        for (int k = 0; k < 8; ++k)
            sum[k] += __expf(acc0[k]) + __expf(acc1[k]);
    }
    // reduce over the 16 lanes of each half (N dimension)
    #pragma unroll
    for (int k = 0; k < 8; ++k) {
        #pragma unroll
        for (int off = 1; off < 16; off <<= 1)
            sum[k] += __shfl_xor(sum[k], off, 32);
    }
    if (lr == 0) {
        const int rbase = i0 + ((lane < 16) ? 0 : 8);
        #pragma unroll
        for (int k = 0; k < 8; ++k) srow[rbase + k] = sum[k];
    }
}

// ---------------------------------------------------------------------------
// Kernel 3: u = x / s  (transposed bf16), and inv_s.
// ---------------------------------------------------------------------------
__global__ void __launch_bounds__(128) k_uprep(const float* __restrict__ x,
                                               const float* __restrict__ srow,
                                               unsigned short* __restrict__ ubT,
                                               float* __restrict__ inv_s) {
    const int row = blockIdx.x;
    const int tid = threadIdx.x;
    const float rs = 1.0f / srow[row];
    const float* xr = x + (size_t)row * D_DIM;
    #pragma unroll
    for (int t = 0; t < 4; ++t) {
        const int c = tid + t * 128;
        ubT[(size_t)c * N_ROWS + row] = f2bf(xr[c] * rs);
    }
    if (tid == 0) inv_s[row] = rs;
}

// ---------------------------------------------------------------------------
// Kernel 4: fused flash pass. Block = 8 waves, 16 output rows.
//   Phase A: recompute S tile (16 x 256) with WMMA, exp -> LDS (bf16 A layout)
//   Phase B: accX += E@x,  accU += E@u  (WMMA, B from transposed bf16 arrays)
//   Epilogue: y = x - 0.1*(inv_s*accX + accU), LayerNorm, store.
// ---------------------------------------------------------------------------
__global__ void __launch_bounds__(256) k_main(const float* __restrict__ x,
                                              const unsigned short* __restrict__ xn,
                                              const unsigned short* __restrict__ xbT,
                                              const unsigned short* __restrict__ ubT,
                                              const float* __restrict__ inv_s,
                                              const float* __restrict__ gamma,
                                              const float* __restrict__ beta,
                                              float* __restrict__ out) {
    __shared__ unsigned short ldsE[16 * JTP];
    __shared__ float ldsSum[16 * 8];
    __shared__ float ldsSq[16 * 8];
    __shared__ float ldsMean[16];
    __shared__ float ldsRstd[16];

    const int lane  = threadIdx.x & 31;
    const int wave  = threadIdx.x >> 5;          // 0..7
    const int lr    = lane & 15;
    const int h8    = (lane < 16) ? 0 : 8;
    const int abase = (lane < 16) ? 0 : 16;
    const int kboff = (lane < 16) ? 0 : 16;
    const int i0    = blockIdx.x * 16;

    const unsigned short* arow = xn + (size_t)(i0 + lr) * D_DIM;

    v8f accX[4]; v8f accU[4];
    #pragma unroll
    for (int t = 0; t < 4; ++t) {
        accX[t] = (v8f){0.f,0.f,0.f,0.f,0.f,0.f,0.f,0.f};
        accU[t] = (v8f){0.f,0.f,0.f,0.f,0.f,0.f,0.f,0.f};
    }

    for (int j = 0; j < N_ROWS; j += JT) {
        __syncthreads();   // previous Phase B finished reading ldsE
        // ---- Phase A: this wave computes two 16x16 S tiles of the 16xJT strip
        #pragma unroll
        for (int t = 0; t < 2; ++t) {
            const int ct = (2 * wave + t) * 16;              // local col base
            const unsigned short* brow = xn + (size_t)(j + ct + lr) * D_DIM;
            v8f sacc = {0.f,0.f,0.f,0.f,0.f,0.f,0.f,0.f};
            #pragma unroll
            for (int kk = 0; kk < 16; ++kk) {
                const int kb = kk * 32;
                VHalf16 a, b;
                a.q[0] = *(const uint4*)(arow + kb + abase);
                a.q[1] = *(const uint4*)(arow + kb + abase + 8);
                b.q[0] = *(const uint4*)(brow + kb + kboff);
                b.q[1] = *(const uint4*)(brow + kb + kboff + 8);
                sacc = __builtin_amdgcn_wmma_f32_16x16x32_bf16(false, a.v, false, b.v,
                                                               (short)0, sacc, false, false);
            }
            #pragma unroll
            for (int k = 0; k < 8; ++k)
                ldsE[(k + h8) * JTP + ct + lr] = f2bf(__expf(sacc[k]));
        }
        __syncthreads();
        // ---- Phase B: E (from LDS, A layout) x {x^T, u^T} (contiguous B loads)
        #pragma unroll
        for (int kc = 0; kc < JT / 32; ++kc) {
            VHalf16 ae;
            const unsigned short* erow = &ldsE[lr * JTP + kc * 32];
            ae.q[0] = *(const uint4*)(erow + abase);
            ae.q[1] = *(const uint4*)(erow + abase + 8);
            #pragma unroll
            for (int t = 0; t < 4; ++t) {
                const int c = wave * 64 + t * 16 + lr;       // output column
                const size_t boff = (size_t)c * N_ROWS + (size_t)j + kc * 32 + kboff;
                VHalf16 bx, bu;
                bx.q[0] = *(const uint4*)(xbT + boff);
                bx.q[1] = *(const uint4*)(xbT + boff + 8);
                bu.q[0] = *(const uint4*)(ubT + boff);
                bu.q[1] = *(const uint4*)(ubT + boff + 8);
                accX[t] = __builtin_amdgcn_wmma_f32_16x16x32_bf16(false, ae.v, false, bx.v,
                                                                  (short)0, accX[t], false, false);
                accU[t] = __builtin_amdgcn_wmma_f32_16x16x32_bf16(false, ae.v, false, bu.v,
                                                                  (short)0, accU[t], false, false);
            }
        }
    }

    // ---- Epilogue: y then LayerNorm (rows split across lanes/waves)
    float invs[8];
    #pragma unroll
    for (int k = 0; k < 8; ++k) invs[k] = inv_s[i0 + k + h8];

    float psum[8], psq[8];
    #pragma unroll
    for (int k = 0; k < 8; ++k) { psum[k] = 0.f; psq[k] = 0.f; }

    #pragma unroll
    for (int t = 0; t < 4; ++t) {
        const int c = wave * 64 + t * 16 + lr;
        #pragma unroll
        for (int k = 0; k < 8; ++k) {
            const int row = i0 + k + h8;
            const float xv = x[(size_t)row * D_DIM + c];
            const float yv = xv - SCALE_F * (invs[k] * accX[t][k] + accU[t][k]);
            accX[t][k] = yv;
            psum[k] += yv;
            psq[k]  += yv * yv;
        }
    }
    #pragma unroll
    for (int k = 0; k < 8; ++k) {
        #pragma unroll
        for (int off = 1; off < 16; off <<= 1) {
            psum[k] += __shfl_xor(psum[k], off, 32);
            psq[k]  += __shfl_xor(psq[k],  off, 32);
        }
    }
    if (lr == 0) {
        #pragma unroll
        for (int k = 0; k < 8; ++k) {
            ldsSum[(k + h8) * 8 + wave] = psum[k];
            ldsSq[(k + h8) * 8 + wave]  = psq[k];
        }
    }
    __syncthreads();
    if (threadIdx.x < 16) {
        const int r = threadIdx.x;
        float s = 0.f, q = 0.f;
        #pragma unroll
        for (int w = 0; w < 8; ++w) { s += ldsSum[r * 8 + w]; q += ldsSq[r * 8 + w]; }
        const float mean = s * (1.0f / D_DIM);
        const float var  = q * (1.0f / D_DIM) - mean * mean;
        ldsMean[r] = mean;
        ldsRstd[r] = rsqrtf(var + LN_EPS);
    }
    __syncthreads();
    #pragma unroll
    for (int t = 0; t < 4; ++t) {
        const int c = wave * 64 + t * 16 + lr;
        const float g = gamma[c];
        const float b = beta[c];
        #pragma unroll
        for (int k = 0; k < 8; ++k) {
            const int row = i0 + k + h8;
            const float m  = ldsMean[k + h8];
            const float rs = ldsRstd[k + h8];
            out[(size_t)row * D_DIM + c] = (accX[t][k] - m) * rs * g + b;
        }
    }
}

// ---------------------------------------------------------------------------
extern "C" void kernel_launch(void* const* d_in, const int* in_sizes, int n_in,
                              void* d_out, int out_size, void* d_ws, size_t ws_size,
                              hipStream_t stream) {
    (void)in_sizes; (void)n_in; (void)out_size; (void)ws_size;
    const float* x     = (const float*)d_in[0];
    const float* gamma = (const float*)d_in[1];
    const float* beta  = (const float*)d_in[2];
    float* out = (float*)d_out;

    // workspace layout (needs ~24.1 MB)
    char* ws = (char*)d_ws;
    const size_t bfMat = (size_t)N_ROWS * D_DIM * sizeof(unsigned short);  // 8 MiB
    unsigned short* xn    = (unsigned short*)(ws);
    unsigned short* xbT   = (unsigned short*)(ws + bfMat);
    unsigned short* ubT   = (unsigned short*)(ws + 2 * bfMat);
    float*          srow  = (float*)(ws + 3 * bfMat);
    float*          inv_s = (float*)(ws + 3 * bfMat + (size_t)N_ROWS * sizeof(float));

    k_prep  <<<N_ROWS,      128, 0, stream>>>(x, xn, xbT);
    k_rowsum<<<N_ROWS / 64, 128, 0, stream>>>(xn, srow);
    k_uprep <<<N_ROWS,      128, 0, stream>>>(x, srow, ubT, inv_s);
    k_main  <<<N_ROWS / 16, 256, 0, stream>>>(x, xn, xbT, ubT, inv_s, gamma, beta, out);
}
